// MinibatchDiscrimination_13408887898456
// MI455X (gfx1250) — matrix operand order repaired and checked by reading, not compiled, not run on hardware
//
#include <hip/hip_runtime.h>

typedef __attribute__((ext_vector_type(16))) _Float16 v16h;
typedef __attribute__((ext_vector_type(8)))  float    v8f;

constexpr int NB_KERNELS = 32;
constexpr int KERNEL_DIM = 16;
constexpr int INPUT_DIM  = 1024;
constexpr int BATCH      = 1024;
constexpr int ACT_COLS   = NB_KERNELS * KERNEL_DIM;   // 512
constexpr int OUT_COLS   = INPUT_DIM + NB_KERNELS;    // 1056

// ---------------------------------------------------------------------------
// Kernel 1: act[b, k*16+m] = sum_d x[b,d] * W[k,d,m]
// One wave32 computes one 16x16 tile: 16 batch rows x (all 16 m's of one k).
// Split-precision f16 WMMA (hi/lo) accumulating in f32 for ~fp32 accuracy.
// ---------------------------------------------------------------------------
__global__ __launch_bounds__(128) void mbd_gemm(const float* __restrict__ x,
                                                const float* __restrict__ W,
                                                float* __restrict__ act) {
  const int lane = threadIdx.x & 31;
  const int wave = threadIdx.x >> 5;
  const int bb   = blockIdx.x * 16;          // batch-row tile base
  const int k    = blockIdx.y * 4 + wave;    // kernel index
  const int h    = lane >> 4;                // lane half (0/1)
  const int l    = lane & 15;                // lane within half

  // A: this lane supplies row (bb+l) of x. Per k-step of 32:
  //   elems 0..7  -> x[row, d0 +      h*8 + i]
  //   elems 8..15 -> x[row, d0 + 16 + h*8 + i]
  const float* xrow = x + (size_t)(bb + l) * INPUT_DIM;
  // B: this lane supplies column n=l of W_k (stride KERNEL_DIM between d's).
  //   elem e<8 -> W[k, d0 + h*8 + e, l]; elem e>=8 -> W[k, d0 + 16 + h*8 + (e-8), l]
  const float* wcol = W + (size_t)k * (INPUT_DIM * KERNEL_DIM) + l;

  v8f acc = {};   // f32 accumulator, zero-init

  for (int d0 = 0; d0 < INPUT_DIM; d0 += 32) {
    // ---- load A chunk (16 floats, two contiguous groups of 8) ----
    float af[16];
    const float4* p0 = reinterpret_cast<const float4*>(xrow + d0 + h * 8);
    const float4* p1 = reinterpret_cast<const float4*>(xrow + d0 + 16 + h * 8);
    float4 u0 = p0[0], u1 = p0[1], u2 = p1[0], u3 = p1[1];
    af[0] = u0.x; af[1] = u0.y; af[2]  = u0.z; af[3]  = u0.w;
    af[4] = u1.x; af[5] = u1.y; af[6]  = u1.z; af[7]  = u1.w;
    af[8] = u2.x; af[9] = u2.y; af[10] = u2.z; af[11] = u2.w;
    af[12] = u3.x; af[13] = u3.y; af[14] = u3.z; af[15] = u3.w;

    // ---- load B chunk (16 floats, strided by KERNEL_DIM) ----
    float bf[16];
    const float* q = wcol + (size_t)d0 * KERNEL_DIM;
#pragma unroll
    for (int i = 0; i < 16; ++i) bf[i] = q[(i < 8 ? h * 8 + i : 16 + h * 8 + (i - 8)) * KERNEL_DIM];

    // ---- split into f16 hi/lo fragments ----
    v16h ahi, alo, bhi, blo;
#pragma unroll
    for (int i = 0; i < 16; ++i) {
      _Float16 xh = (_Float16)af[i];
      ahi[i] = xh;
      alo[i] = (_Float16)(af[i] - (float)xh);
      _Float16 wh = (_Float16)bf[i];
      bhi[i] = wh;
      blo[i] = (_Float16)(bf[i] - (float)wh);
    }

    // acc += alo*bhi + ahi*blo + ahi*bhi   (error terms first)
    acc = __builtin_amdgcn_wmma_f32_16x16x32_f16(false, alo, false, bhi,
                                                 (short)0, acc, false, false);
    acc = __builtin_amdgcn_wmma_f32_16x16x32_f16(false, ahi, false, blo,
                                                 (short)0, acc, false, false);
    acc = __builtin_amdgcn_wmma_f32_16x16x32_f16(false, ahi, false, bhi,
                                                 (short)0, acc, false, false);
  }

  // C/D layout: lane L, VGPR r -> (M = r + 8*h, N = l)
#pragma unroll
  for (int r = 0; r < 8; ++r)
    act[(size_t)(bb + r + 8 * h) * ACT_COLS + k * KERNEL_DIM + l] = acc[r];
}

// ---------------------------------------------------------------------------
// Kernel 2: features[b,k] = sum_c exp(-sum_m |act[b,k,m]-act[c,k,m]|)
// Block = (256 batch rows, one k); c-rows tiled through 16 KB of LDS.
// Writes directly into out[:, 1024 + k].
// ---------------------------------------------------------------------------
__global__ __launch_bounds__(256) void mbd_pairwise(const float* __restrict__ act,
                                                    float* __restrict__ out) {
  const int k = blockIdx.y;
  const int b = blockIdx.x * 256 + threadIdx.x;

  __shared__ float4 tile[256 * 4];   // 256 rows x 16 floats

  const float4* arow = reinterpret_cast<const float4*>(
      act + (size_t)b * ACT_COLS + k * KERNEL_DIM);
  float4 m0 = arow[0], m1 = arow[1], m2 = arow[2], m3 = arow[3];

  float sum = 0.0f;
  for (int c0 = 0; c0 < BATCH; c0 += 256) {
    __syncthreads();
    const float4* crow = reinterpret_cast<const float4*>(
        act + (size_t)(c0 + threadIdx.x) * ACT_COLS + k * KERNEL_DIM);
    tile[threadIdx.x * 4 + 0] = crow[0];
    tile[threadIdx.x * 4 + 1] = crow[1];
    tile[threadIdx.x * 4 + 2] = crow[2];
    tile[threadIdx.x * 4 + 3] = crow[3];
    __syncthreads();

#pragma unroll 4
    for (int c = 0; c < 256; ++c) {
      float4 t0 = tile[c * 4 + 0];
      float4 t1 = tile[c * 4 + 1];
      float4 t2 = tile[c * 4 + 2];
      float4 t3 = tile[c * 4 + 3];
      float d = fabsf(m0.x - t0.x) + fabsf(m0.y - t0.y) +
                fabsf(m0.z - t0.z) + fabsf(m0.w - t0.w) +
                fabsf(m1.x - t1.x) + fabsf(m1.y - t1.y) +
                fabsf(m1.z - t1.z) + fabsf(m1.w - t1.w) +
                fabsf(m2.x - t2.x) + fabsf(m2.y - t2.y) +
                fabsf(m2.z - t2.z) + fabsf(m2.w - t2.w) +
                fabsf(m3.x - t3.x) + fabsf(m3.y - t3.y) +
                fabsf(m3.z - t3.z) + fabsf(m3.w - t3.w);
      sum += __expf(-d);
    }
  }

  out[(size_t)b * OUT_COLS + INPUT_DIM + k] = sum;
}

// ---------------------------------------------------------------------------
// Kernel 3: out[:, 0:1024] = x   (float4 copy; 1056-float row stride is
// 16B aligned)
// ---------------------------------------------------------------------------
__global__ __launch_bounds__(256) void mbd_copyx(const float* __restrict__ x,
                                                 float* __restrict__ out) {
  const int i   = blockIdx.x * 256 + threadIdx.x;   // float4 index
  const int row = i >> 8;                            // 256 float4 per row
  const int col = i & 255;
  float4 v = reinterpret_cast<const float4*>(x)[i];
  reinterpret_cast<float4*>(out + (size_t)row * OUT_COLS)[col] = v;
}

// ---------------------------------------------------------------------------
extern "C" void kernel_launch(void* const* d_in, const int* in_sizes, int n_in,
                              void* d_out, int out_size, void* d_ws, size_t ws_size,
                              hipStream_t stream) {
  const float* x = (const float*)d_in[0];   // (1024, 1024)
  const float* W = (const float*)d_in[1];   // (32, 1024, 16)
  float* out = (float*)d_out;               // (1024, 1056)
  float* act = (float*)d_ws;                // (1024, 512) scratch = 2 MB

  // GEMM: 64 row-tiles x 32 kernels, 4 waves (4 k's) per 128-thread block.
  mbd_gemm<<<dim3(BATCH / 16, NB_KERNELS / 4), 128, 0, stream>>>(x, W, act);

  // Pairwise L1 + exp reduction, writes features into out[:, 1024:1056].
  mbd_pairwise<<<dim3(BATCH / 256, NB_KERNELS), 256, 0, stream>>>(act, out);

  // Copy x into out[:, 0:1024].
  mbd_copyx<<<(BATCH * INPUT_DIM / 4) / 256, 256, 0, stream>>>(x, out);
}
